// ProbAttention_56599079027327
// MI455X (gfx1250) — matrix operand order repaired and checked by reading, not compiled, not run on hardware
//
#include <hip/hip_runtime.h>
#include <hip/hip_bf16.h>

// B=512, T=200; L-dimension padded to TP=208 (13*16); stored rows padded to
// TPV=224 (7*32) so the PV GEMM's K loop needs no bounds checks at all.
#define NB    512
#define TT    200
#define TP    208
#define TPV   224
#define HH    256
#define RT_N  13     // 16-row tiles covering TP
#define RT_NV 14     // 16-row tiles covering TPV

typedef __attribute__((ext_vector_type(16))) __bf16 v16bf;
typedef __attribute__((ext_vector_type(8)))  float  v8f;

union Pack16 { uint4 u[2]; v16bf v; __bf16 h[16]; };
union Pack8  { uint4 u;    __bf16 h[8]; };

__device__ __forceinline__ v8f wmma_bf16(v16bf a, v16bf b, v8f c) {
  return __builtin_amdgcn_wmma_f32_16x16x32_bf16(false, a, false, b, (short)0, c,
                                                 false, false);
}

// Build a 16x32 bf16 A-operand (contiguous k = 16*half + s order, same as load_b_bf16) from a contiguous bf16 row.
__device__ __forceinline__ v16bf load_a_bf16(const __bf16* row, int k0, int half) {
  Pack16 p;
  p.u[0] = *(const uint4*)(row + k0 + half * 16);       // K = k0+h16 .. +7
  p.u[1] = *(const uint4*)(row + k0 + half * 16 + 8);   // K = k0+h16+8 .. +15
  return p.v;
}
// Build a 32x16 bf16 B-operand: 16 contiguous K-values per lane.
__device__ __forceinline__ v16bf load_b_bf16(const __bf16* row, int kb) {
  Pack16 p;
  p.u[0] = *(const uint4*)(row + kb);
  p.u[1] = *(const uint4*)(row + kb + 8);
  return p.v;
}

// ---------------------------------------------------------------------------
// K0: transpose+convert both weight matrices to bf16: Wt[n][k] = bf16(W[k][n])
// ---------------------------------------------------------------------------
__global__ __launch_bounds__(256) void transpose_w_kernel(
    const float* __restrict__ WL, const float* __restrict__ WV,
    __bf16* __restrict__ WtL, __bf16* __restrict__ WtV)
{
  const int n = blockIdx.x, k = threadIdx.x;
  WtL[(size_t)n * HH + k] = (__bf16)WL[(size_t)k * HH + n];
  WtV[(size_t)n * HH + k] = (__bf16)WV[(size_t)k * HH + n];
}

// ---------------------------------------------------------------------------
// K1/K2: C = X @ W + bias in bf16. One wave per 16x16 tile, K=256 -> 8 WMMA.
// TRANS=false: row-major out [NB][TPV][HH] (for S).
// TRANS=true : transposed out [NB][HH][TPV] (for V), pad rows zeroed.
// All loads unconditional: A-row pointer clamped (garbage only reaches pad
// C-rows, which are stored as zero).
// ---------------------------------------------------------------------------
template <bool TRANS>
__global__ __launch_bounds__(32) void gemm_proj_kernel(
    const float*  __restrict__ X,     // [NB*TT, HH]
    const __bf16* __restrict__ Wt,    // [HH][HH] transposed bf16
    const float*  __restrict__ bias,  // [HH]
    __bf16* __restrict__ Outp)
{
  const int mt    = blockIdx.x;           // 0 .. NB*RT_NV-1
  const int batch = mt / RT_NV;
  const int r0    = (mt % RT_NV) * 16;    // row offset within padded TPV
  const int lane  = threadIdx.x;
  const int half  = lane >> 4;
  const int l15   = lane & 15;
  const int col   = blockIdx.y * 16 + l15;

  const int grow  = r0 + l15;
  const int crow  = grow < TT ? grow : (TT - 1);        // clamped, no branch
  const float*  xrow = X + ((size_t)batch * TT + crow) * HH;
  const __bf16* wrow = Wt + (size_t)col * HH;

  v8f acc = {};
  for (int kk = 0; kk < 8; ++kk) {
    const int k0 = kk * 32;
    const float4 fa0 = *(const float4*)(xrow + k0 + half * 16);
    const float4 fa1 = *(const float4*)(xrow + k0 + half * 16 + 4);
    const float4 fb0 = *(const float4*)(xrow + k0 + half * 16 + 8);
    const float4 fb1 = *(const float4*)(xrow + k0 + half * 16 + 12);
    v16bf a;
    a[0]=(__bf16)fa0.x; a[1]=(__bf16)fa0.y; a[2]=(__bf16)fa0.z; a[3]=(__bf16)fa0.w;
    a[4]=(__bf16)fa1.x; a[5]=(__bf16)fa1.y; a[6]=(__bf16)fa1.z; a[7]=(__bf16)fa1.w;
    a[8]=(__bf16)fb0.x; a[9]=(__bf16)fb0.y; a[10]=(__bf16)fb0.z; a[11]=(__bf16)fb0.w;
    a[12]=(__bf16)fb1.x; a[13]=(__bf16)fb1.y; a[14]=(__bf16)fb1.z; a[15]=(__bf16)fb1.w;
    const v16bf bm = load_b_bf16(wrow, k0 + half * 16);
    if (kk < 7)  // speculative prefetch -> global_prefetch_b8
      __builtin_prefetch(wrow + k0 + 32 + half * 16, 0, 1);
    acc = wmma_bf16(a, bm, acc);
  }
  const float bv = bias[col];
  if (TRANS) {
    Pack8 ps;
#pragma unroll
    for (int v = 0; v < 8; ++v) {
      const int gr = r0 + half * 8 + v;                 // contiguous in v
      ps.h[v] = (__bf16)((gr < TT) ? (acc[v] + bv) : 0.f);
    }
    *(uint4*)(Outp + ((size_t)batch * HH + col) * TPV + r0 + half * 8) = ps.u;
  } else {
#pragma unroll
    for (int v = 0; v < 8; ++v) {
      const int gr = r0 + half * 8 + v;
      const float val = (gr < TT) ? (acc[v] + bv) : 0.f;
      Outp[((size_t)batch * TPV + gr) * HH + col] = (__bf16)val;
    }
  }
}

// ---------------------------------------------------------------------------
// K3: per batch: dL_i = ||S_i||^2, sum_det = 0.5*((sum dL)^2 - ||L||_F^2).
// ---------------------------------------------------------------------------
__global__ __launch_bounds__(256) void dpp_stats_kernel(
    const __bf16* __restrict__ S,   // [NB][TPV][HH]
    float* __restrict__ dLout,      // [NB][TPV]
    float* __restrict__ sumdet)     // [NB]
{
  const int b    = blockIdx.x;
  const int tid  = threadIdx.x;
  const int lane = tid & 31;
  const int wid  = tid >> 5;
  const int half = lane >> 4;
  const int l15  = lane & 15;

  __shared__ float s_dl[TP];
  __shared__ float s_part[8];

  const __bf16* Sb = S + (size_t)b * TPV * HH;

  if (tid < TP) {
    const __bf16* row = Sb + (size_t)tid * HH;
    float s = 0.f;
    for (int k = 0; k < HH; k += 8) {
      Pack8 p; p.u = *(const uint4*)(row + k);
#pragma unroll
      for (int e = 0; e < 8; ++e) { const float v = (float)p.h[e]; s += v * v; }
    }
    s_dl[tid] = s;
    dLout[(size_t)b * TPV + tid] = s;
  }
  __syncthreads();

  float frob = 0.f;
  for (int t = wid; t < RT_N * RT_N; t += 8) {
    const int ti = t / RT_N, tj = t % RT_N;
    const __bf16* arow = Sb + (size_t)(ti * 16 + l15) * HH;
    const __bf16* brow = Sb + (size_t)(tj * 16 + l15) * HH;  // row of S = col of S^T
    v8f c = {};
    for (int kk = 0; kk < 8; ++kk) {
      const int k0 = kk * 32;
      c = wmma_bf16(load_a_bf16(arow, k0, half),
                    load_b_bf16(brow, k0 + half * 16), c);
    }
#pragma unroll
    for (int v = 0; v < 8; ++v) frob += c[v] * c[v];
  }
  for (int m = 16; m; m >>= 1) frob += __shfl_xor(frob, m, 32);
  if (lane == 0) s_part[wid] = frob;
  __syncthreads();
  if (tid == 0) {
    float F = 0.f;
    for (int i = 0; i < 8; ++i) F += s_part[i];
    float sdl = 0.f;
    for (int i = 0; i < TP; ++i) sdl += s_dl[i];
    sumdet[b] = 0.5f * (sdl * sdl - F);
  }
}

// ---------------------------------------------------------------------------
// K4: per (batch, 16-row block): L block via WMMA -> P -> row stats ->
// softmax weights (bf16 in LDS) -> W @ Vt via WMMA -> residual + LayerNorm.
// ---------------------------------------------------------------------------
__global__ __launch_bounds__(256) void dpp_attn_kernel(
    const float*  __restrict__ X,      // [NB][TT][HH]
    const __bf16* __restrict__ S,      // [NB][TPV][HH]
    const __bf16* __restrict__ Vt,     // [NB][HH][TPV]  (rows j zero-padded)
    const float*  __restrict__ dL,     // [NB][TPV]
    const float*  __restrict__ sumdet, // [NB]
    const float*  __restrict__ ln_g,
    const float*  __restrict__ ln_b,
    float* __restrict__ Out)           // [NB][TT][HH]
{
  const int blk  = blockIdx.x;
  const int b    = blk / RT_N;
  const int r0   = (blk % RT_N) * 16;
  const int tid  = threadIdx.x;
  const int lane = tid & 31;
  const int wid  = tid >> 5;
  const int half = lane >> 4;
  const int l15  = lane & 15;

  __shared__ float  sP[16][TP];       // L block, then P block
  __shared__ __bf16 sW[16][TPV];      // softmax weights, K padded to 224
  __shared__ float  sOut[16][HH];     // pre-LN hidden + residual
  __shared__ float  s_dl[TP];
  __shared__ float  sM[16], sD[16], sZ[16];
  __shared__ float  s_sd;

  const __bf16* Sb  = S  + (size_t)b * TPV * HH;
  const __bf16* Vtb = Vt + (size_t)b * HH * TPV;

  if (tid < TP) s_dl[tid] = dL[(size_t)b * TPV + tid];
  if (tid == 0) s_sd = sumdet[b] + 1e-5f;
  __syncthreads();

  // ---- L row-block: fixed 16 A-rows, 13 column tiles over 8 waves
  const __bf16* arow = Sb + (size_t)(r0 + l15) * HH;
  for (int t = wid; t < RT_N; t += 8) {
    const __bf16* brow = Sb + (size_t)(t * 16 + l15) * HH;
    v8f c = {};
    for (int kk = 0; kk < 8; ++kk) {
      const int k0 = kk * 32;
      c = wmma_bf16(load_a_bf16(arow, k0, half),
                    load_b_bf16(brow, k0 + half * 16), c);
    }
#pragma unroll
    for (int v = 0; v < 8; ++v) sP[v + half * 8][t * 16 + l15] = c[v];
  }
  __syncthreads();

  // ---- P = exp(-det / (sum_det+eps)) elementwise
  const float inv_sd = 1.0f / s_sd;
  for (int idx = tid; idx < 16 * TP; idx += 256) {
    const int i = idx / TP, j = idx % TP;
    float p = 0.f;
    if (j < TT) {
      const float l   = sP[i][j];
      const float det = s_dl[r0 + i] * s_dl[j] - l * l;
      p = __expf(-det * inv_sd);
    }
    sP[i][j] = p;
  }
  __syncthreads();

  // ---- row stats: rowsum(P), max off-diag, D_ii, m
  for (int rr = 0; rr < 2; ++rr) {
    const int i  = wid * 2 + rr;
    const int dj = r0 + i;
    float rs = 0.f, mx = -3.4e38f;
    for (int j = lane; j < TT; j += 32) {
      const float p = sP[i][j];
      rs += p;
      if (j != dj) mx = fmaxf(mx, p);
    }
    for (int m = 16; m; m >>= 1) {
      rs += __shfl_xor(rs, m, 32);
      mx = fmaxf(mx, __shfl_xor(mx, m, 32));
    }
    if (lane == 0) {
      const float pii = (dj < TT) ? sP[i][dj] : 0.f;
      const float Dii = rs - pii;
      sD[i] = Dii;
      sM[i] = fmaxf(mx, Dii);
    }
  }
  __syncthreads();

  // ---- Z_i (deterministic shuffle reduction)
  for (int rr = 0; rr < 2; ++rr) {
    const int i  = wid * 2 + rr;
    const int dj = r0 + i;
    const float mrow = sM[i];
    float z = 0.f;
    for (int j = lane; j < TT; j += 32) {
      const float d = (j == dj) ? sD[i] : sP[i][j];
      z += __expf(d - mrow);
    }
    for (int m = 16; m; m >>= 1) z += __shfl_xor(z, m, 32);
    if (lane == 0) sZ[i] = z;
  }
  __syncthreads();

  // ---- softmax numerator weights as bf16 (invalid rows forced to 0)
  for (int idx = tid; idx < 16 * TPV; idx += 256) {   // 14 elems/thread
    const int i = idx / TPV, j = idx % TPV;
    float w = 0.f;
    if (j < TT && (r0 + i) < TT) {
      const float d = (j == r0 + i) ? sD[i] : sP[i][j];
      w = __expf(d - sM[i]);
    }
    sW[i][j] = (__bf16)w;
  }
  __syncthreads();

  // ---- hidden = (W @ V) / Z + residual; 2 output column tiles per wave
  for (int ct = 0; ct < 2; ++ct) {
    const int coln = (wid * 2 + ct) * 16 + l15;
    const __bf16* vrow = Vtb + (size_t)coln * TPV;    // contiguous over j
    v8f c = {};
    for (int kk = 0; kk < 7; ++kk) {                  // K = 224, no guards
      const int k0 = kk * 32;
      c = wmma_bf16(load_a_bf16(&sW[l15][0], k0, half),
                    load_b_bf16(vrow, k0 + half * 16), c);
    }
#pragma unroll
    for (int v = 0; v < 8; ++v) {
      const int row = v + half * 8;
      const int gr  = r0 + row;
      const float z = sZ[row];
      float hv = (z > 0.f) ? (c[v] / z) : 0.f;
      if (gr < TT) hv += X[((size_t)b * TT + gr) * HH + coln];
      sOut[row][coln] = hv;
    }
  }
  __syncthreads();

  // ---- LayerNorm per row (wave handles 2 rows, 8 cols/lane)
  for (int rr = 0; rr < 2; ++rr) {
    const int row = wid * 2 + rr;
    const int gr  = r0 + row;
    float s = 0.f, ss = 0.f;
#pragma unroll
    for (int e = 0; e < 8; ++e) {
      const float v = sOut[row][lane * 8 + e];
      s += v; ss += v * v;
    }
    for (int m = 16; m; m >>= 1) {
      s  += __shfl_xor(s, m, 32);
      ss += __shfl_xor(ss, m, 32);
    }
    const float mean = s * (1.f / HH);
    const float var  = ss * (1.f / HH) - mean * mean;
    const float rinv = rsqrtf(var + 1e-12f);
    if (gr < TT) {
      float* orow = Out + ((size_t)b * TT + gr) * HH;
#pragma unroll
      for (int e = 0; e < 8; ++e) {
        const int cidx = lane * 8 + e;
        orow[cidx] = (sOut[row][cidx] - mean) * rinv * ln_g[cidx] + ln_b[cidx];
      }
    }
  }
}

// ---------------------------------------------------------------------------
extern "C" void kernel_launch(void* const* d_in, const int* in_sizes, int n_in,
                              void* d_out, int out_size, void* d_ws, size_t ws_size,
                              hipStream_t stream) {
  const float* x    = (const float*)d_in[0];
  // d_in[1] attention_mask: unused by the forward pass
  const float* W_L  = (const float*)d_in[2];
  const float* b_L  = (const float*)d_in[3];
  const float* W_v  = (const float*)d_in[4];
  const float* b_v  = (const float*)d_in[5];
  const float* ln_g = (const float*)d_in[6];
  const float* ln_b = (const float*)d_in[7];
  float* out = (float*)d_out;

  // workspace layout (all offsets 256B-aligned):
  char* ws = (char*)d_ws;
  const size_t szS = (size_t)NB * TPV * HH * sizeof(__bf16);   // ~58.7 MB
  __bf16* wsS   = (__bf16*)ws;
  __bf16* wsVt  = (__bf16*)(ws + szS);
  __bf16* wsWtL = (__bf16*)(ws + 2 * szS);
  __bf16* wsWtV = (__bf16*)(ws + 2 * szS + (size_t)HH * HH * sizeof(__bf16));
  float*  wsdL  = (float*)(ws + 2 * szS + 2 * (size_t)HH * HH * sizeof(__bf16));
  float*  wssd  = (float*)((char*)wsdL + (size_t)NB * TPV * sizeof(float));

  transpose_w_kernel<<<HH, HH, 0, stream>>>(W_L, W_v, wsWtL, wsWtV);

  dim3 gProj(NB * RT_NV, HH / 16);
  gemm_proj_kernel<false><<<gProj, 32, 0, stream>>>(x, wsWtL, b_L, wsS);
  gemm_proj_kernel<true ><<<gProj, 32, 0, stream>>>(x, wsWtV, b_v, wsVt);

  dpp_stats_kernel<<<NB, 256, 0, stream>>>(wsS, wsdL, wssd);
  dpp_attn_kernel<<<NB * RT_N, 256, 0, stream>>>(x, wsS, wsVt, wsdL, wssd,
                                                 ln_g, ln_b, out);
}
